// VCModel_78829829751248
// MI455X (gfx1250) — compile-verified, matrix operand order checked by reading, not run on hardware
//
#include <hip/hip_runtime.h>

// ---------------------------------------------------------------------------
// bf16 helpers (ushort storage; __bf16 only for WMMA fragments)
// ---------------------------------------------------------------------------
typedef __attribute__((ext_vector_type(16))) __bf16          v16bf;
typedef __attribute__((ext_vector_type(8)))  float           v8f;
typedef __attribute__((ext_vector_type(8)))  unsigned short  v8u16;
typedef __attribute__((ext_vector_type(4)))  int             v4i;

typedef __attribute__((address_space(1))) v4i v4i_g;   // global int4
typedef __attribute__((address_space(3))) v4i v4i_l;   // LDS int4

#if defined(__has_builtin)
#if __has_builtin(__builtin_amdgcn_global_load_async_to_lds_b128)
#define HAVE_ASYNC_LDS 1
#endif
#endif

__device__ __forceinline__ unsigned short f2bf(float f) {
    unsigned u = __float_as_uint(f);
    u = u + 0x7FFFu + ((u >> 16) & 1u);   // round-to-nearest-even
    return (unsigned short)(u >> 16);
}
__device__ __forceinline__ float bf2f(unsigned short h) {
    return __uint_as_float(((unsigned)h) << 16);
}

__device__ __forceinline__ void wait_async_all() {
#if defined(HAVE_ASYNC_LDS)
#if __has_builtin(__builtin_amdgcn_s_wait_asynccnt)
    __builtin_amdgcn_s_wait_asynccnt(0);
#else
    asm volatile("s_wait_asynccnt 0x0" ::: "memory");
#endif
#endif
}

// ---------------------------------------------------------------------------
// Generic bf16 WMMA GEMM: C[M,N] = act(A[M,K] * B[K,N] + bias)
//   A:  [M,K] row-major bf16 (lda elements), M must be a multiple of 128
//   Bt: [N,K] bf16 (weights pre-transposed: contiguous 128-bit along K)
// block = 256 threads = 8 waves; block tile 128(M) x 64(N); K step 32.
// Each wave owns 16(M) x 64(N): one A fragment reused across 4 WMMAs.
// ALIGNED=1 (K%32==0 && N%64==0): branch-free staging, async LDS DMA when
// available (GLOBAL_LOAD_ASYNC_TO_LDS_B128, ASYNCcnt-tracked).
// ---------------------------------------------------------------------------
template <int ALIGNED>
__global__ void __launch_bounds__(256)
gemm_bf16_wmma(const unsigned short* __restrict__ A, int lda,
               const unsigned short* __restrict__ Bt,
               const float* __restrict__ bias,
               void* __restrict__ C, int ldc, int cdtype, int relu,
               long long M, int N, int K)
{
    __shared__ unsigned short Alds[128][40];   // 128 M x 32 K (+pad), 80B rows
    __shared__ unsigned short Blds[64][40];    //  64 N x 32 K (+pad)

    const int tid   = threadIdx.x;
    const int wave  = tid >> 5;          // 0..7 -> M sub-tile
    const int lane  = tid & 31;
    const int l16   = lane & 15;
    const int khalf = lane >> 4;

    const long long m0 = (long long)blockIdx.y * 128;
    const int       n0 = blockIdx.x * 64;

    // staging coordinates (fixed per thread)
    const int ar0 = tid >> 2;            // A rows handled: ar0 and ar0+64
    const int ac  = (tid & 3) * 8;       // K chunk offset
    const int br  = tid >> 2;            // Bt row (N)
    const long long aBase0 = (m0 + ar0)      * (long long)lda + ac;
    const long long aBase1 = (m0 + ar0 + 64) * (long long)lda + ac;
    const long long bBase  = (long long)(n0 + br) * K + ac;

    v8f acc[4];
    #pragma unroll
    for (int nt = 0; nt < 4; ++nt) acc[nt] = v8f{0.f,0.f,0.f,0.f,0.f,0.f,0.f,0.f};

    const v8u16 zero8 = {0,0,0,0,0,0,0,0};

    for (int k0 = 0; k0 < K; k0 += 32) {
        if (ALIGNED) {
#if defined(HAVE_ASYNC_LDS)
            // ---- async DMA global -> LDS (no VGPR roundtrip) ----
            __builtin_amdgcn_global_load_async_to_lds_b128(
                (v4i_g*)(void*)&A[aBase0 + k0],
                (v4i_l*)&Alds[ar0][ac], 0, 0);
            __builtin_amdgcn_global_load_async_to_lds_b128(
                (v4i_g*)(void*)&A[aBase1 + k0],
                (v4i_l*)&Alds[ar0 + 64][ac], 0, 0);
            __builtin_amdgcn_global_load_async_to_lds_b128(
                (v4i_g*)(void*)&Bt[bBase + k0],
                (v4i_l*)&Blds[br][ac], 0, 0);
            wait_async_all();
#else
            *(v8u16*)&Alds[ar0][ac]      = *(const v8u16*)&A[aBase0 + k0];
            *(v8u16*)&Alds[ar0 + 64][ac] = *(const v8u16*)&A[aBase1 + k0];
            *(v8u16*)&Blds[br][ac]       = *(const v8u16*)&Bt[bBase + k0];
#endif
            if (k0 + 32 < K) {   // prefetch next K tiles -> global_prefetch_b8
                __builtin_prefetch(&A[aBase0 + k0 + 32], 0, 1);
                __builtin_prefetch(&Bt[bBase + k0 + 32], 0, 1);
            }
        } else {
            // ---- masked staging for K/N tails ----
            int kk = k0 + ac;
            v8u16 x0 = zero8, x1 = zero8, xb = zero8;
            if (kk < K) {
                x0 = *(const v8u16*)&A[aBase0 + k0];
                x1 = *(const v8u16*)&A[aBase1 + k0];
                if (n0 + br < N) xb = *(const v8u16*)&Bt[bBase + k0];
            }
            *(v8u16*)&Alds[ar0][ac]      = x0;
            *(v8u16*)&Alds[ar0 + 64][ac] = x1;
            *(v8u16*)&Blds[br][ac]       = xb;
        }
        __syncthreads();

        // ---- fragments: two contiguous b128 LDS loads each ----
        union Frag { v16bf v; v8u16 h[2]; };
        Frag a;
        a.h[0] = *(const v8u16*)&Alds[wave * 16 + l16][khalf * 8];
        a.h[1] = *(const v8u16*)&Alds[wave * 16 + l16][16 + khalf * 8];
        #pragma unroll
        for (int nt = 0; nt < 4; ++nt) {
            Frag b;
            b.h[0] = *(const v8u16*)&Blds[nt * 16 + l16][khalf * 8];
            b.h[1] = *(const v8u16*)&Blds[nt * 16 + l16][16 + khalf * 8];
            acc[nt] = __builtin_amdgcn_wmma_f32_16x16x32_bf16(
                          false, a.v, false, b.v, (short)0, acc[nt], false, false);
        }
        __syncthreads();
    }

    // ---- epilogue: bias + relu; C layout: VGPR r -> M = r + 8*khalf ----
    #pragma unroll
    for (int nt = 0; nt < 4; ++nt) {
        const int col = n0 + nt * 16 + l16;
        if (ALIGNED || col < N) {
            const float bv = bias ? bias[col] : 0.0f;
            #pragma unroll
            for (int r = 0; r < 8; ++r) {
                long long row = m0 + wave * 16 + khalf * 8 + r;
                float v = acc[nt][r] + bv;
                if (relu) v = v > 0.f ? v : 0.f;
                if (cdtype == 0) ((float*)C)[row * (long long)ldc + col] = v;
                else ((unsigned short*)C)[row * (long long)ldc + col] = f2bf(v);
            }
        }
    }
}

// ---------------------------------------------------------------------------
// fp32 -> bf16 convert (plain, and transposing for weights: [K,N] -> [N,K])
// ---------------------------------------------------------------------------
__global__ void cvt_f32_bf16_kernel(const float* __restrict__ src,
                                    unsigned short* __restrict__ dst,
                                    long long n)
{
    long long i = (long long)blockIdx.x * 256 + threadIdx.x;
    if (i < n) dst[i] = f2bf(src[i]);
}

__global__ void cvt_transpose_bf16_kernel(const float* __restrict__ src, /*[K,N]*/
                                          unsigned short* __restrict__ dst, /*[N,K]*/
                                          int K, int N)
{
    long long i = (long long)blockIdx.x * 256 + threadIdx.x;
    if (i >= (long long)K * N) return;
    int n = (int)(i / K), k = (int)(i - (long long)n * K);
    dst[i] = f2bf(src[(long long)k * N + n]);
}

__global__ void fill_zero_kernel(float* __restrict__ p, long long n)
{
    long long i = (long long)blockIdx.x * 256 + threadIdx.x;
    if (i < n) p[i] = 0.0f;
}

// ---------------------------------------------------------------------------
// LayerNorm over last dim n (rows = gridDim.x), dual bf16 / fp32 output
// ---------------------------------------------------------------------------
__global__ void __launch_bounds__(256)
layernorm_kernel(const float* __restrict__ x,
                 const float* __restrict__ g, const float* __restrict__ b,
                 unsigned short* __restrict__ ybf, float* __restrict__ yf32,
                 int n)
{
    const long long row = blockIdx.x;
    const float* xr = x + row * (long long)n;
    __shared__ float rs[256], rq[256];
    float s = 0.f, q = 0.f;
    for (int i = threadIdx.x; i < n; i += 256) {
        float v = xr[i]; s += v; q += v * v;
    }
    rs[threadIdx.x] = s; rq[threadIdx.x] = q;
    __syncthreads();
    for (int off = 128; off > 0; off >>= 1) {
        if (threadIdx.x < off) {
            rs[threadIdx.x] += rs[threadIdx.x + off];
            rq[threadIdx.x] += rq[threadIdx.x + off];
        }
        __syncthreads();
    }
    const float mean = rs[0] / n;
    const float var  = rq[0] / n - mean * mean;
    const float inv  = rsqrtf(var + 1e-5f);
    for (int i = threadIdx.x; i < n; i += 256) {
        float v = (xr[i] - mean) * inv * g[i] + b[i];
        if (ybf)  ybf[row * (long long)n + i]  = f2bf(v);
        if (yf32) yf32[row * (long long)n + i] = v;
    }
}

// ---------------------------------------------------------------------------
// im2col for conv1d k=3, SAME, per-1024-frame sequence boundaries
// in [rows,512] bf16 -> out [rows,1536] bf16
// ---------------------------------------------------------------------------
__global__ void im2col3_kernel(const unsigned short* __restrict__ in,
                               unsigned short* __restrict__ out,
                               long long rows)
{
    long long idx = (long long)blockIdx.x * 256 + threadIdx.x;
    long long total = rows * 1536;
    if (idx >= total) return;
    long long r = idx / 1536;
    int c = (int)(idx - r * 1536);
    int d = c / 512, cin = c - d * 512;
    int t = (int)(r & 1023);
    int tt = t + d - 1;
    unsigned short v = 0;
    if (tt >= 0 && tt < 1024) v = in[(r + d - 1) * 512 + cin];
    out[idx] = v;
}

// ---------------------------------------------------------------------------
// Embedding gathers (ref rows first: 65536, then src rows: 16384)
// ---------------------------------------------------------------------------
__global__ void embed_key_kernel(const int* __restrict__ ref_e, const int* __restrict__ ref_p,
                                 const int* __restrict__ src_e, const int* __restrict__ src_p,
                                 const float* __restrict__ Etab, const float* __restrict__ Ptab,
                                 unsigned short* __restrict__ out /*[81920,512]*/)
{
    long long idx = (long long)blockIdx.x * 256 + threadIdx.x;   // rows*256
    long long r = idx >> 8;  int c = (int)(idx & 255);
    if (r >= 81920) return;
    int ei, pi;
    if (r < 65536) { ei = ref_e[r]; pi = ref_p[r]; }
    else           { ei = src_e[r - 65536]; pi = src_p[r - 65536]; }
    float v = (c < 128) ? Etab[(long long)ei * 128 + c]
                        : Ptab[(long long)pi * 128 + (c - 128)];
    out[r * 512 + c] = f2bf(v);
}

__global__ void embed_val_kernel(const int* __restrict__ ref_e, const int* __restrict__ src_e,
                                 const float* __restrict__ Etab,
                                 unsigned short* __restrict__ out /*[81920,384]*/)
{
    long long idx = (long long)blockIdx.x * 256 + threadIdx.x;   // rows*128
    long long r = idx >> 7;  int c = (int)(idx & 127);
    if (r >= 81920) return;
    int ei = (r < 65536) ? ref_e[r] : src_e[r - 65536];
    out[r * 384 + c] = f2bf(Etab[(long long)ei * 128 + c]);
}

__global__ void embed_dec_kernel(const int* __restrict__ src_e, const int* __restrict__ src_p,
                                 const float* __restrict__ Etab, const float* __restrict__ Ptab,
                                 unsigned short* __restrict__ out /*[16384,768]*/)
{
    long long idx = (long long)blockIdx.x * 256 + threadIdx.x;   // rows*256
    long long r = idx >> 8;  int c = (int)(idx & 255);
    if (r >= 16384) return;
    float v = (c < 128) ? Etab[(long long)src_e[r] * 128 + c]
                        : Ptab[(long long)src_p[r] * 128 + (c - 128)];
    out[r * 768 + 512 + c] = f2bf(v);
}

// ---------------------------------------------------------------------------
// Assemble kv bank: kv[b,j] = src[b, s+j] (j<l) else ref[j>>10, b, j&1023]
// ---------------------------------------------------------------------------
__global__ void __launch_bounds__(256)
kv_assemble_kernel(const float* __restrict__ keyf /*[81920,32]*/,
                   const float* __restrict__ valf /*[81920,512]*/,
                   const int* __restrict__ src_p, const int* __restrict__ ref_p,
                   const int* __restrict__ s_ptr, const int* __restrict__ l_ptr,
                   float* __restrict__ out_key, unsigned short* __restrict__ kvkey_bf,
                   float* __restrict__ out_val, unsigned short* __restrict__ kvval_bf,
                   float* __restrict__ out_pitch)
{
    const int b = blockIdx.x >> 12;
    const int j = blockIdx.x & 4095;
    const int s = s_ptr[0], l = l_ptr[0];
    long long srow;
    int pitch;
    if (j < l) {
        srow  = (long long)(64 + b) * 1024 + (s + j);           // src rows live at 65536+
        pitch = src_p[b * 1024 + s + j];
    } else {
        srow  = (long long)((j >> 10) * 16 + b) * 1024 + (j & 1023);
        pitch = ref_p[srow];
    }
    const long long kvrow = (long long)blockIdx.x;              // b*4096 + j
    if (threadIdx.x < 32) {
        float v = keyf[srow * 32 + threadIdx.x];
        out_key[kvrow * 32 + threadIdx.x]  = v;
        kvkey_bf[kvrow * 32 + threadIdx.x] = f2bf(v);
    }
    for (int c = threadIdx.x; c < 512; c += 256) {
        float v = valf[srow * 512 + c];
        out_val[kvrow * 512 + c]  = v;
        kvval_bf[kvrow * 512 + c] = f2bf(v);
    }
    if (threadIdx.x == 0) out_pitch[kvrow] = (float)pitch;
}

// ---------------------------------------------------------------------------
// Hard attention: argmax_j (q . k_j) per query; write idx + one-hot attn.
// ---------------------------------------------------------------------------
__global__ void __launch_bounds__(256)
argmax_attn_kernel(const float* __restrict__ Q  /*[16384,32]*/,
                   const float* __restrict__ Kp /*[65536,32]*/,
                   int* __restrict__ idx_out,
                   float* __restrict__ attn /*[16,1024,4096]*/)
{
    const int q = blockIdx.x;
    const int b = q >> 10;
    __shared__ float qv[32];
    if (threadIdx.x < 32) qv[threadIdx.x] = Q[(long long)q * 32 + threadIdx.x];
    __syncthreads();
    float best = -3.4e38f; int bi = 0x7fffffff;
    for (int j = threadIdx.x; j < 4096; j += 256) {
        const float* kr = Kp + ((long long)b * 4096 + j) * 32;
        float d = 0.f;
        #pragma unroll
        for (int c = 0; c < 32; ++c) d += qv[c] * kr[c];
        if (d > best) { best = d; bi = j; }
    }
    __shared__ float rbest[256]; __shared__ int ridx[256];
    rbest[threadIdx.x] = best; ridx[threadIdx.x] = bi;
    __syncthreads();
    for (int off = 128; off > 0; off >>= 1) {
        if (threadIdx.x < off) {
            float ob = rbest[threadIdx.x + off]; int oi = ridx[threadIdx.x + off];
            if (ob > rbest[threadIdx.x] ||
                (ob == rbest[threadIdx.x] && oi < ridx[threadIdx.x])) {
                rbest[threadIdx.x] = ob; ridx[threadIdx.x] = oi;
            }
        }
        __syncthreads();
    }
    if (threadIdx.x == 0) {
        idx_out[q] = ridx[0];
        attn[(long long)q * 4096 + ridx[0]] = 1.0f;
    }
}

// ---------------------------------------------------------------------------
// Gather selected value rows -> bf16 [16384,512]
// ---------------------------------------------------------------------------
__global__ void gather_val_kernel(const float* __restrict__ vproj /*[65536,512]*/,
                                  const int* __restrict__ idx,
                                  unsigned short* __restrict__ out /*[16384,512]*/)
{
    long long i = (long long)blockIdx.x * 256 + threadIdx.x;   // 16384*512
    if (i >= 16384LL * 512) return;
    long long row = i >> 9; int c = (int)(i & 511);
    int b = (int)(row >> 10);
    int j = idx[row];
    out[i] = f2bf(vproj[((long long)b * 4096 + j) * 512 + c]);
}

// ---------------------------------------------------------------------------
// Host launcher
// ---------------------------------------------------------------------------
enum {
    I_SRC_E = 0, I_SRC_P, I_SRC_W2V2, I_SRC_MEL,
    I_REF_E, I_REF_P, I_REF_W2V2, I_REF_MEL,
    I_S, I_L,
    I_EEMB, I_PEMB, I_W2V2_W, I_W2V2_B, I_MEL_W, I_MEL_B,
    I_K_W1, I_K_B1, I_K_G1, I_K_BE1, I_K_CW, I_K_CB, I_K_G2, I_K_BE2,
    I_K_W2, I_K_B2, I_K_G3, I_K_BE3,
    I_V_W1, I_V_B1, I_V_G1, I_V_BE1, I_V_CW, I_V_CB, I_V_G2, I_V_BE2,
    I_V_W2, I_V_B2, I_V_G3, I_V_BE3,
    I_D_W1, I_D_B1, I_D_G1, I_D_BE1, I_D_CW, I_D_CB, I_D_G2, I_D_BE2,
    I_D_W2, I_D_B2, I_D_G3, I_D_BE3,
    I_D_W3, I_D_B3,
    I_A_WQ, I_A_BQ, I_A_WK, I_A_BK, I_A_WV, I_A_BV, I_A_WO, I_A_BO
};

extern "C" void kernel_launch(void* const* d_in, const int* in_sizes, int n_in,
                              void* d_out, int out_size, void* d_ws, size_t ws_size,
                              hipStream_t stream)
{
    (void)in_sizes; (void)n_in; (void)out_size; (void)ws_size;

    const long long ROWS     = 81920;   // 4*16*1024 ref rows + 16*1024 src rows
    const long long ROWS_REF = 65536;
    const long long ROWS_SRC = 16384;
    const long long KVROWS   = 65536;   // 16 * 4096

    // ---- workspace bump allocator ----
    char* ws = (char*)d_ws;
    size_t off = 0;
    auto carve = [&](size_t bytes) -> char* {
        char* p = ws + off;
        off += (bytes + 255) & ~(size_t)255;
        return p;
    };
    auto bfb = [&](long long n) { return (unsigned short*)carve((size_t)n * 2); };
    auto f32b = [&](long long n) { return (float*)carve((size_t)n * 4); };

    // bf16 weights, transposed to [N,K]
    unsigned short* bw_w2v2 = bfb(256 * 256);
    unsigned short* bw_mel  = bfb(80 * 256);
    unsigned short* bw_kw1  = bfb(512 * 512);
    unsigned short* bw_kcw  = bfb(1536 * 512);
    unsigned short* bw_kw2  = bfb(512 * 32);
    unsigned short* bw_vw1  = bfb(384 * 512);
    unsigned short* bw_vcw  = bfb(1536 * 512);
    unsigned short* bw_vw2  = bfb(512 * 512);
    unsigned short* bw_dw1  = bfb(768 * 512);
    unsigned short* bw_dcw  = bfb(1536 * 512);
    unsigned short* bw_dw2  = bfb(512 * 512);
    unsigned short* bw_dw3  = bfb(512 * 80);
    unsigned short* bw_awq  = bfb(32 * 32);
    unsigned short* bw_awk  = bfb(32 * 32);
    unsigned short* bw_awv  = bfb(512 * 512);
    unsigned short* bw_awo  = bfb(512 * 512);

    // activations
    unsigned short* W2V2_BF = bfb(ROWS * 256);
    unsigned short* MEL_BF  = bfb(ROWS * 80);
    unsigned short* ACT_IN  = bfb(ROWS * 512);    // block input (512 or 384 stride)
    unsigned short* ACT_H   = bfb(ROWS * 512);    // post-LN bf16
    unsigned short* ACT_A   = bfb(ROWS * 1536);   // im2col
    float*          ACT_F32 = f32b(ROWS * 512);   // GEMM out before LN
    float*          KEY_F32 = f32b(ROWS * 32);
    unsigned short* KEY_BF  = bfb(ROWS * 32);
    float*          VAL_F32 = f32b(ROWS * 512);
    unsigned short* KVK_BF  = bfb(KVROWS * 32);
    unsigned short* KVV_BF  = bfb(KVROWS * 512);
    float*          QF      = f32b(ROWS_SRC * 32);
    float*          KF      = f32b(KVROWS * 32);
    int*            IDX     = (int*)carve(ROWS_SRC * 4);
    float*          VPROJ   = f32b(KVROWS * 512);
    unsigned short* TGT_BF  = bfb(ROWS_SRC * 512);
    unsigned short* IND_BF  = bfb(ROWS_SRC * 768);

    // output regions (float, concatenated in return order)
    float* out = (float*)d_out;
    float* O_MEL   = out;
    float* O_KEY   = O_MEL + 16384LL * 80;
    float* O_VAL   = O_KEY + 16LL * 4096 * 32;
    float* O_PITCH = O_VAL + 16LL * 4096 * 512;
    float* O_ATTN  = O_PITCH + 16LL * 4096;

    auto gb = [](long long n) { return (unsigned)((n + 255) / 256); };
    auto cvt = [&](int idx, unsigned short* dst, long long n) {
        cvt_f32_bf16_kernel<<<gb(n), 256, 0, stream>>>((const float*)d_in[idx], dst, n);
    };
    auto cvtT = [&](int idx, unsigned short* dst, int K, int N) {   // [K,N] -> [N,K]
        cvt_transpose_bf16_kernel<<<gb((long long)K * N), 256, 0, stream>>>(
            (const float*)d_in[idx], dst, K, N);
    };
    auto gemm = [&](const unsigned short* A, int lda,
                    const unsigned short* Bt, const float* bias,
                    void* C, int ldc, int cdt, int relu,
                    long long M, int N, int K) {
        dim3 g((unsigned)((N + 63) / 64), (unsigned)((M + 127) / 128));
        if ((K % 32 == 0) && (N % 64 == 0))
            gemm_bf16_wmma<1><<<g, 256, 0, stream>>>(A, lda, Bt, bias, C, ldc,
                                                     cdt, relu, M, N, K);
        else
            gemm_bf16_wmma<0><<<g, 256, 0, stream>>>(A, lda, Bt, bias, C, ldc,
                                                     cdt, relu, M, N, K);
    };
    auto ln = [&](const float* x, int gidx, int bidx,
                  unsigned short* ybf, float* yf32, long long rows, int n) {
        layernorm_kernel<<<(unsigned)rows, 256, 0, stream>>>(
            x, (const float*)d_in[gidx], (const float*)d_in[bidx], ybf, yf32, n);
    };

    // ---- weight conversion (transpose [K,N] -> [N,K]) ----
    cvtT(I_W2V2_W, bw_w2v2, 256, 256);  cvtT(I_MEL_W, bw_mel, 80, 256);
    cvtT(I_K_W1, bw_kw1, 512, 512);     cvtT(I_K_CW, bw_kcw, 1536, 512);
    cvtT(I_K_W2, bw_kw2, 512, 32);
    cvtT(I_V_W1, bw_vw1, 384, 512);     cvtT(I_V_CW, bw_vcw, 1536, 512);
    cvtT(I_V_W2, bw_vw2, 512, 512);
    cvtT(I_D_W1, bw_dw1, 768, 512);     cvtT(I_D_CW, bw_dcw, 1536, 512);
    cvtT(I_D_W2, bw_dw2, 512, 512);     cvtT(I_D_W3, bw_dw3, 512, 80);
    cvtT(I_A_WQ, bw_awq, 32, 32);       cvtT(I_A_WK, bw_awk, 32, 32);
    cvtT(I_A_WV, bw_awv, 512, 512);     cvtT(I_A_WO, bw_awo, 512, 512);

    // ---- input conversion (ref rows first, then src rows) ----
    cvt(I_REF_W2V2, W2V2_BF, ROWS_REF * 256);
    cvt(I_SRC_W2V2, W2V2_BF + ROWS_REF * 256, ROWS_SRC * 256);
    cvt(I_REF_MEL, MEL_BF, ROWS_REF * 80);
    cvt(I_SRC_MEL, MEL_BF + ROWS_REF * 80, ROWS_SRC * 80);

    // ============================ KEY PATH ============================
    gemm(W2V2_BF, 256, bw_w2v2, (const float*)d_in[I_W2V2_B],
         ACT_IN + 256, 512, 1, 0, ROWS, 256, 256);
    embed_key_kernel<<<gb(ROWS * 256), 256, 0, stream>>>(
        (const int*)d_in[I_REF_E], (const int*)d_in[I_REF_P],
        (const int*)d_in[I_SRC_E], (const int*)d_in[I_SRC_P],
        (const float*)d_in[I_EEMB], (const float*)d_in[I_PEMB], ACT_IN);
    gemm(ACT_IN, 512, bw_kw1, (const float*)d_in[I_K_B1],
         ACT_F32, 512, 0, 1, ROWS, 512, 512);
    ln(ACT_F32, I_K_G1, I_K_BE1, ACT_H, nullptr, ROWS, 512);
    im2col3_kernel<<<gb(ROWS * 1536), 256, 0, stream>>>(ACT_H, ACT_A, ROWS);
    gemm(ACT_A, 1536, bw_kcw, (const float*)d_in[I_K_CB],
         ACT_F32, 512, 0, 1, ROWS, 512, 1536);
    ln(ACT_F32, I_K_G2, I_K_BE2, ACT_H, nullptr, ROWS, 512);
    gemm(ACT_H, 512, bw_kw2, (const float*)d_in[I_K_B2],
         ACT_F32, 32, 0, 1, ROWS, 32, 512);
    ln(ACT_F32, I_K_G3, I_K_BE3, KEY_BF, KEY_F32, ROWS, 32);

    // ============================ VALUE PATH ==========================
    gemm(MEL_BF, 80, bw_mel, (const float*)d_in[I_MEL_B],
         ACT_IN + 128, 384, 1, 0, ROWS, 256, 80);
    embed_val_kernel<<<gb(ROWS * 128), 256, 0, stream>>>(
        (const int*)d_in[I_REF_E], (const int*)d_in[I_SRC_E],
        (const float*)d_in[I_EEMB], ACT_IN);
    gemm(ACT_IN, 384, bw_vw1, (const float*)d_in[I_V_B1],
         ACT_F32, 512, 0, 1, ROWS, 512, 384);
    ln(ACT_F32, I_V_G1, I_V_BE1, ACT_H, nullptr, ROWS, 512);
    im2col3_kernel<<<gb(ROWS * 1536), 256, 0, stream>>>(ACT_H, ACT_A, ROWS);
    gemm(ACT_A, 1536, bw_vcw, (const float*)d_in[I_V_CB],
         ACT_F32, 512, 0, 1, ROWS, 512, 1536);
    ln(ACT_F32, I_V_G2, I_V_BE2, ACT_H, nullptr, ROWS, 512);
    gemm(ACT_H, 512, bw_vw2, (const float*)d_in[I_V_B2],
         ACT_F32, 512, 0, 1, ROWS, 512, 512);
    ln(ACT_F32, I_V_G3, I_V_BE3, nullptr, VAL_F32, ROWS, 512);

    // ============================ KV ASSEMBLY =========================
    kv_assemble_kernel<<<16 * 4096, 256, 0, stream>>>(
        KEY_F32, VAL_F32,
        (const int*)d_in[I_SRC_P], (const int*)d_in[I_REF_P],
        (const int*)d_in[I_S], (const int*)d_in[I_L],
        O_KEY, KVK_BF, O_VAL, KVV_BF, O_PITCH);

    // ============================ ATTENTION ===========================
    gemm(KEY_BF + ROWS_REF * 32, 32, bw_awq, (const float*)d_in[I_A_BQ],
         QF, 32, 0, 0, ROWS_SRC, 32, 32);
    gemm(KVK_BF, 32, bw_awk, (const float*)d_in[I_A_BK],
         KF, 32, 0, 0, KVROWS, 32, 32);
    fill_zero_kernel<<<gb(16LL * 1024 * 4096), 256, 0, stream>>>(
        O_ATTN, 16LL * 1024 * 4096);
    argmax_attn_kernel<<<(unsigned)ROWS_SRC, 256, 0, stream>>>(QF, KF, IDX, O_ATTN);
    gemm(KVV_BF, 512, bw_awv, (const float*)d_in[I_A_BV],
         VPROJ, 512, 0, 0, KVROWS, 512, 512);
    gather_val_kernel<<<gb(ROWS_SRC * 512), 256, 0, stream>>>(VPROJ, IDX, TGT_BF);
    gemm(TGT_BF, 512, bw_awo, (const float*)d_in[I_A_BO],
         IND_BF, 768, 1, 0, ROWS_SRC, 512, 512);   // -> cols [0,512) of in_d
    embed_dec_kernel<<<gb(ROWS_SRC * 256), 256, 0, stream>>>(
        (const int*)d_in[I_SRC_E], (const int*)d_in[I_SRC_P],
        (const float*)d_in[I_EEMB], (const float*)d_in[I_PEMB], IND_BF);

    // ============================ DECODE ==============================
    gemm(IND_BF, 768, bw_dw1, (const float*)d_in[I_D_B1],
         ACT_F32, 512, 0, 1, ROWS_SRC, 512, 768);
    ln(ACT_F32, I_D_G1, I_D_BE1, ACT_H, nullptr, ROWS_SRC, 512);
    im2col3_kernel<<<gb(ROWS_SRC * 1536), 256, 0, stream>>>(ACT_H, ACT_A, ROWS_SRC);
    gemm(ACT_A, 1536, bw_dcw, (const float*)d_in[I_D_CB],
         ACT_F32, 512, 0, 1, ROWS_SRC, 512, 1536);
    ln(ACT_F32, I_D_G2, I_D_BE2, ACT_H, nullptr, ROWS_SRC, 512);
    gemm(ACT_H, 512, bw_dw2, (const float*)d_in[I_D_B2],
         ACT_F32, 512, 0, 1, ROWS_SRC, 512, 512);
    ln(ACT_F32, I_D_G3, I_D_BE3, ACT_H, nullptr, ROWS_SRC, 512);
    gemm(ACT_H, 512, bw_dw3, (const float*)d_in[I_D_B3],
         O_MEL, 80, 0, 0, ROWS_SRC, 80, 512);
}
